// SparseMoE_18296560681213
// MI455X (gfx1250) — matrix operand order repaired and checked by reading, not compiled, not run on hardware
//
#include <hip/hip_runtime.h>
#include <hip/hip_bf16.h>

// ---------------------------------------------------------------------------
// SparseMoE (noisy top-2 router + grouped expert FFN) for gfx1250 (MI455X).
// - Router: noisy top-2 softmax, per-expert token lists (atomic append).
// - Weights pre-converted once per launch to bf16 in WMMA *fragment order*
//   (conv_w), so GEMM B-fragments are raw 2x global_load_b128 per lane.
// - Grouped GEMM via v_wmma_f32_16x16x32_bf16, f32 accumulate.
// - GEMM2 A-tile staged with global_load_async_to_lds_b128 (ASYNCcnt path).
// - h staged in d_ws as bf16; combine via global atomicAdd(f32).
// ---------------------------------------------------------------------------

typedef __attribute__((ext_vector_type(16))) __bf16 v16bf;
typedef __attribute__((ext_vector_type(2)))  __bf16 bf16x2;
typedef __attribute__((ext_vector_type(8)))  float  v8f;

#define T_TOK 8192   // B*S
#define DIM   1024   // D
#define HID   4096   // H = 4D
#define NEXP  8      // E

// workspace layout (bytes)
#define WS_COUNTS  ((size_t)0)
#define WS_OFFSETS ((size_t)256)
#define WS_TOKIDX  ((size_t)512)
#define WS_TOKGATE (WS_TOKIDX + (size_t)NEXP * T_TOK * 4)
#define WS_W1F     ((size_t)1 << 20)                       // 64 MB frag bf16
#define WS_W2F     (WS_W1F + (size_t)NEXP * DIM * HID * 2) // 64 MB frag bf16
#define WS_HBUF    (WS_W2F + (size_t)NEXP * HID * DIM * 2) // 128 MB bf16

// ---- helpers ---------------------------------------------------------------

__device__ __forceinline__ unsigned short f2bf(float f) {
    union { __bf16 b; unsigned short u; } c;
    c.b = (__bf16)f;                 // native v_cvt (RNE)
    return c.u;
}

__device__ __forceinline__ unsigned pack2bf(float a, float b) {
    union { bf16x2 v; unsigned u; } c;
    c.v.x = (__bf16)a;               // fuses to v_cvt_pk_bf16_f32
    c.v.y = (__bf16)b;
    return c.u;
}

union AFrag { v16bf v; unsigned u[8]; uint4 q[2]; };
union CFrag { v8f   v; float    f[8]; };

// ---- kernel 0: zero output + counters --------------------------------------

__global__ void moe_zero(float4* __restrict__ out4, int* __restrict__ counts) {
    size_t i = (size_t)blockIdx.x * blockDim.x + threadIdx.x;
    const size_t n4 = (size_t)T_TOK * DIM / 4;
    const size_t stride = (size_t)gridDim.x * blockDim.x;
    float4 z = {0.f, 0.f, 0.f, 0.f};
    for (size_t k = i; k < n4; k += stride) out4[k] = z;
    if (i < NEXP) counts[i] = 0;
}

// ---- kernel 1: noisy top-2 router ------------------------------------------

__global__ void moe_router(const float* __restrict__ logits,
                           const float* __restrict__ noise,
                           int* __restrict__ counts,
                           int* __restrict__ tokIdx,
                           float* __restrict__ tokGate) {
    int t = blockIdx.x * blockDim.x + threadIdx.x;
    if (t >= T_TOK) return;
    float v[NEXP];
#pragma unroll
    for (int e = 0; e < NEXP; ++e) {
        float l  = logits[t * NEXP + e];
        float nz = noise[t * NEXP + e];
        float sp = log1pf(expf(-fabsf(l))) + fmaxf(l, 0.0f);  // stable softplus
        v[e] = l + nz * sp;
    }
    int i1 = 0;                                 // first index wins ties
#pragma unroll
    for (int e = 1; e < NEXP; ++e) if (v[e] > v[i1]) i1 = e;
    int i2 = (i1 == 0) ? 1 : 0;
#pragma unroll
    for (int e = 0; e < NEXP; ++e) if (e != i1 && v[e] > v[i2]) i2 = e;

    float e2 = expf(v[i2] - v[i1]);             // v[i1] >= v[i2]
    float inv = 1.0f / (1.0f + e2);

    int s1 = atomicAdd(&counts[i1], 1);
    tokIdx[i1 * T_TOK + s1]  = t;
    tokGate[i1 * T_TOK + s1] = inv;
    int s2 = atomicAdd(&counts[i2], 1);
    tokIdx[i2 * T_TOK + s2]  = t;
    tokGate[i2 * T_TOK + s2] = e2 * inv;
}

// ---- kernel 2: exclusive prefix over 8 counts ------------------------------

__global__ void moe_prefix(const int* __restrict__ counts, int* __restrict__ offsets) {
    if (threadIdx.x == 0 && blockIdx.x == 0) {
        int s = 0;
        for (int e = 0; e < NEXP; ++e) { offsets[e] = s; s += counts[e]; }
    }
}

// ---- kernel 2b: convert W [E][K][N] fp32 -> bf16 in WMMA fragment order ----
// Fragment tile = 32(K) x 16(N); tile id = (e*KT + kt)*NT + nt; 256 dwords per
// tile; lane l owns dwords [l*8, l*8+8): n = nt*16 + (l&15),
// K = kt*32 + 16*(l>>4) + {2i, 2i+1}.  One wave per tile; reads are 64B
// coalesced row segments, writes are contiguous 1KB per wave.

__global__ __launch_bounds__(256) void conv_w(const float* __restrict__ W,
                                              unsigned* __restrict__ dst,
                                              int K, int N) {
    const int tile = blockIdx.x * 8 + (threadIdx.x >> 5);
    const int lane = threadIdx.x & 31;
    const int NT = N >> 4;
    const int nt = tile % NT;
    const int ktE = tile / NT;                 // e*KT + kt
    const int n  = nt * 16 + (lane & 15);
    const int k0 = (ktE % (K >> 5)) * 32 + 16 * (lane >> 4);
    const int e  = ktE / (K >> 5);
    const float* __restrict__ col = W + (size_t)e * K * N + n;
    unsigned* __restrict__ o = dst + (size_t)tile * 256 + lane * 8;
#pragma unroll
    for (int i = 0; i < 8; ++i) {
        float f0 = col[(size_t)(k0 + 2 * i)     * N];
        float f1 = col[(size_t)(k0 + 2 * i + 1) * N];
        o[i] = pack2bf(f0, f1);
    }
}

// ---- WMMA tile machinery ---------------------------------------------------
// Workgroup: 256 threads = 8 waves; 64x64 out tile = 16 subtiles (16x16);
// wave w owns mi = w%4, ni in {2*(w/4), 2*(w/4)+1}.
// A staged in LDS row-major bf16 [m][k], stride LDS_K; per-lane A fragment is
// two contiguous 16B runs: (m, 16*hf..) and (m, 16*hf+8..) -> 2x ds_load_b128.
// B fragments load straight from the fragment-ordered weight buffer:
// 2x global_load_b128 per lane per fragment.

#define LDS_K 40  // padded K stride (halfwords); rows stay 16B-aligned

__device__ __forceinline__ void build_a_frag(AFrag& a, const unsigned short* As,
                                             int mi, int mr, int hf) {
    int m = mi * 16 + mr;
    a.q[0] = *(const uint4*)&As[m * LDS_K + 16 * hf];
    a.q[1] = *(const uint4*)&As[m * LDS_K + 16 * hf + 8];
}

__device__ __forceinline__ void load_b_frag(AFrag& b, const uint4* __restrict__ p) {
    b.q[0] = p[0];
    b.q[1] = p[1];
}

// ---- kernel 3: h = relu(Xg @ W1f[e] + b1[e]) -------------------------------

__global__ __launch_bounds__(256) void moe_gemm1(
    const float* __restrict__ x, const uint4* __restrict__ w1f,
    const float* __restrict__ b1, const int* __restrict__ counts,
    const int* __restrict__ offsets, const int* __restrict__ tokIdx,
    unsigned short* __restrict__ hbuf) {

    __shared__ __align__(16) unsigned short As[64 * LDS_K];
    __shared__ int rowTok[64];

    const int e = blockIdx.z;
    const int cnt = counts[e];
    const int pairBase = blockIdx.y * 64;
    if (pairBase >= cnt) return;                 // uniform per block
    const int nBase = blockIdx.x * 64;
    const int pbase = offsets[e];
    const int KT = DIM >> 5, NT = HID >> 4;

    const int tid  = threadIdx.x;
    const int lane = tid & 31;
    const int wave = tid >> 5;
    const int hf   = lane >> 4;
    const int mr   = lane & 15;
    const int mi   = wave & 3;
    const int ni0  = (wave >> 2) * 2;

    if (tid < 64) {
        int pr = pairBase + tid;
        rowTok[tid] = tokIdx[e * T_TOK + min(pr, cnt - 1)];  // clamped: no guards
    }
    __syncthreads();

    // K-invariant loader assignments
    const int arow = tid >> 2;                 // A: row 0..63
    const int ac0  = (tid & 3) * 8;            // A: 8-wide k chunk
    const float* __restrict__ xrow = x + (size_t)rowTok[arow] * DIM + ac0;

    // B fragment base: tile (e, kt=0, nt = nBase/16 + ni0), lane slice
    const uint4* __restrict__ wb =
        w1f + ((size_t)e * KT * NT + (nBase >> 4) + ni0) * 64 + lane * 2;
    const size_t wstep = (size_t)NT * 64;      // uint4 per kt step

    v8f acc0 = {0.f,0.f,0.f,0.f,0.f,0.f,0.f,0.f};
    v8f acc1 = {0.f,0.f,0.f,0.f,0.f,0.f,0.f,0.f};

    for (int kBase = 0; kBase < DIM; kBase += 32) {
        // A tile: two b128 loads -> bf16 pack -> one b128 LDS store
        const float4* xv = (const float4*)(xrow + kBase);
        float4 f0 = xv[0], f1 = xv[1];
        uint4 p;
        p.x = pack2bf(f0.x, f0.y); p.y = pack2bf(f0.z, f0.w);
        p.z = pack2bf(f1.x, f1.y); p.w = pack2bf(f1.z, f1.w);
        *(uint4*)&As[arow * LDS_K + ac0] = p;
        __syncthreads();

        AFrag a, b0, b1f;
        build_a_frag(a, As, mi, mr, hf);
        load_b_frag(b0,  wb);
        load_b_frag(b1f, wb + 64);
        if (kBase + 32 < DIM) __builtin_prefetch(wb + wstep, 0, 0);
        wb += wstep;
        acc0 = __builtin_amdgcn_wmma_f32_16x16x32_bf16(false, a.v, false, b0.v,
                                                       (short)0, acc0, false, false);
        acc1 = __builtin_amdgcn_wmma_f32_16x16x32_bf16(false, a.v, false, b1f.v,
                                                       (short)0, acc1, false, false);
        __syncthreads();
    }

    // epilogue: relu(acc + b1) -> hbuf (bf16), compacted pair rows
    CFrag c0; c0.v = acc0;
    CFrag c1; c1.v = acc1;
#pragma unroll
    for (int s = 0; s < 2; ++s) {
        const CFrag& c = s ? c1 : c0;
        int n = nBase + (ni0 + s) * 16 + mr;
        float bias = b1[e * HID + n];
#pragma unroll
        for (int r = 0; r < 8; ++r) {
            int mrow = mi * 16 + 8 * hf + r;
            int pr = pairBase + mrow;
            if (pr < cnt) {
                float v = c.f[r] + bias;
                v = v > 0.0f ? v : 0.0f;
                hbuf[(size_t)(pbase + pr) * HID + n] = f2bf(v);
            }
        }
    }
}

// ---- kernel 4: out += gate * (h @ W2f[e] + b2[e]) --------------------------

__global__ __launch_bounds__(256) void moe_gemm2(
    const unsigned short* __restrict__ hbuf, const uint4* __restrict__ w2f,
    const float* __restrict__ b2, const int* __restrict__ counts,
    const int* __restrict__ offsets, const int* __restrict__ tokIdx,
    const float* __restrict__ tokGate, float* __restrict__ out) {

    __shared__ __align__(16) unsigned short As[64 * LDS_K];
    __shared__ int   rowTok[64];
    __shared__ float rowGate[64];

    const int e = blockIdx.z;
    const int cnt = counts[e];
    const int pairBase = blockIdx.y * 64;
    if (pairBase >= cnt) return;                 // uniform per block
    const int nBase = blockIdx.x * 64;
    const int pbase = offsets[e];
    const int KT = HID >> 5, NT = DIM >> 4;

    const int tid  = threadIdx.x;
    const int lane = tid & 31;
    const int wave = tid >> 5;
    const int hf   = lane >> 4;
    const int mr   = lane & 15;
    const int mi   = wave & 3;
    const int ni0  = (wave >> 2) * 2;

    if (tid < 64) {
        int pr = pairBase + tid;
        bool ok = (pr < cnt);
        rowTok[tid]  = ok ? tokIdx[e * T_TOK + pr]  : 0;
        rowGate[tid] = ok ? tokGate[e * T_TOK + pr] : 0.0f;
    }
    __syncthreads();

    const int arow = tid >> 2;
    const int ac0  = (tid & 3) * 8;
    const int prc  = min(pairBase + arow, cnt - 1);          // clamped: no guards
    const unsigned short* __restrict__ hrow = hbuf + (size_t)(pbase + prc) * HID + ac0;

    // LDS byte address for this thread's A slot: ISA 10.1.1/10.2 — the LDS
    // aperture maps addr[31:0] directly as the wave-relative LDS address, so
    // the low 32 bits of the generic shared pointer are DS-usable.
    const unsigned aLds = (unsigned)(uintptr_t)&As[arow * LDS_K + ac0];

    const uint4* __restrict__ wb =
        w2f + ((size_t)e * KT * NT + (nBase >> 4) + ni0) * 64 + lane * 2;
    const size_t wstep = (size_t)NT * 64;

    v8f acc0 = {0.f,0.f,0.f,0.f,0.f,0.f,0.f,0.f};
    v8f acc1 = {0.f,0.f,0.f,0.f,0.f,0.f,0.f,0.f};

    for (int kBase = 0; kBase < HID; kBase += 32) {
        // A tile: h already bf16 -> async DMA global -> LDS (ASYNCcnt path)
        asm volatile("global_load_async_to_lds_b128 %0, %1, off"
                     :: "v"(aLds), "v"((const void*)(hrow + kBase))
                     : "memory");
        asm volatile("s_wait_asynccnt 0x0" ::: "memory");
        __syncthreads();

        AFrag a, b0, b1f;
        build_a_frag(a, As, mi, mr, hf);
        load_b_frag(b0,  wb);
        load_b_frag(b1f, wb + 64);
        if (kBase + 32 < HID) __builtin_prefetch(wb + wstep, 0, 0);
        wb += wstep;
        acc0 = __builtin_amdgcn_wmma_f32_16x16x32_bf16(false, a.v, false, b0.v,
                                                       (short)0, acc0, false, false);
        acc1 = __builtin_amdgcn_wmma_f32_16x16x32_bf16(false, a.v, false, b1f.v,
                                                       (short)0, acc1, false, false);
        __syncthreads();
    }

    // epilogue: scatter-add gate * (acc + b2) into out[token, n]
    CFrag c0; c0.v = acc0;
    CFrag c1; c1.v = acc1;
#pragma unroll
    for (int s = 0; s < 2; ++s) {
        const CFrag& c = s ? c1 : c0;
        int n = nBase + (ni0 + s) * 16 + mr;
        float bias = b2[e * DIM + n];
#pragma unroll
        for (int r = 0; r < 8; ++r) {
            int mrow = mi * 16 + 8 * hf + r;
            int pr = pairBase + mrow;
            if (pr < cnt) {
                atomicAdd(&out[(size_t)rowTok[mrow] * DIM + n],
                          rowGate[mrow] * (c.f[r] + bias));
            }
        }
    }
}

// ---- host launcher ---------------------------------------------------------

extern "C" void kernel_launch(void* const* d_in, const int* in_sizes, int n_in,
                              void* d_out, int out_size, void* d_ws, size_t ws_size,
                              hipStream_t stream) {
    (void)in_sizes; (void)n_in; (void)out_size; (void)ws_size;
    const float* x      = (const float*)d_in[0];
    const float* logits = (const float*)d_in[1];
    const float* noise  = (const float*)d_in[2];
    const float* W1     = (const float*)d_in[3];
    const float* b1     = (const float*)d_in[4];
    const float* W2     = (const float*)d_in[5];
    const float* b2     = (const float*)d_in[6];
    float* out = (float*)d_out;

    char* ws = (char*)d_ws;
    int*            counts  = (int*)(ws + WS_COUNTS);
    int*            offsets = (int*)(ws + WS_OFFSETS);
    int*            tokIdx  = (int*)(ws + WS_TOKIDX);
    float*          tokGate = (float*)(ws + WS_TOKGATE);
    unsigned*       w1f     = (unsigned*)(ws + WS_W1F);
    unsigned*       w2f     = (unsigned*)(ws + WS_W2F);
    unsigned short* hbuf    = (unsigned short*)(ws + WS_HBUF);

    moe_zero<<<1024, 256, 0, stream>>>((float4*)out, counts);
    moe_router<<<T_TOK / 256, 256, 0, stream>>>(logits, noise, counts, tokIdx, tokGate);
    moe_prefix<<<1, 32, 0, stream>>>(counts, offsets);
    // tiles/8 blocks: E*(K/32)*(N/16)/8
    conv_w<<<NEXP * (DIM / 32) * (HID / 16) / 8, 256, 0, stream>>>(W1, w1f, DIM, HID);
    conv_w<<<NEXP * (HID / 32) * (DIM / 16) / 8, 256, 0, stream>>>(W2, w2f, HID, DIM);
    moe_gemm1<<<dim3(HID / 64, T_TOK / 64, NEXP), 256, 0, stream>>>(
        x, (const uint4*)w1f, b1, counts, offsets, tokIdx, hbuf);
    moe_gemm2<<<dim3(DIM / 64, T_TOK / 64, NEXP), 256, 0, stream>>>(
        hbuf, (const uint4*)w2f, b2, counts, offsets, tokIdx, tokGate, out);
}
